// MultiHeadAttention_2808908611616
// MI455X (gfx1250) — matrix operand order repaired and checked by reading, not compiled, not run on hardware
//
#include <hip/hip_runtime.h>
#include <hip/hip_bf16.h>

#define T_SEQ 4096
#define C_DIM 768
#define H_HEADS 12
#define D_HEAD 64
#define NSUB 4  // n-tiles (16 cols each) per wave in GEMM
#define MSUB 2  // m-tiles (16 rows each) per wave in GEMM

#if __has_builtin(__builtin_amdgcn_global_load_async_to_lds_b128)
#define HAVE_ASYNC_LDS 1
#else
#define HAVE_ASYNC_LDS 0
#endif

#define AS_GLOBAL __attribute__((address_space(1)))
#define AS_LOCAL  __attribute__((address_space(3)))

typedef __attribute__((ext_vector_type(16))) _Float16 v16h;
typedef __attribute__((ext_vector_type(8)))  _Float16 v8h;
typedef __attribute__((ext_vector_type(8)))  float    v8f;
typedef __attribute__((ext_vector_type(4)))  int      v4i;

__device__ __forceinline__ v8f wmma_f16(v16h a, v16h b, v8f c) {
  // D = A(16x32 f16) x B(32x16 f16) + C(16x16 f32)
  return __builtin_amdgcn_wmma_f32_16x16x32_f16(false, a, false, b, (short)0, c,
                                                false, false);
}

// A-operand tile (16 rows x 32 K) from a row-major source (global or LDS).
// Lane l: row = l&15; K = k0 + (l>>4)*16 + 0..15.
__device__ __forceinline__ v16h load_a_tile(const _Float16* __restrict__ base,
                                            int stride, int k0) {
  const int lane = threadIdx.x & 31;
  const _Float16* p = base + (size_t)(lane & 15) * stride + k0 + ((lane >> 4) << 4);
  v8h lo = *(const v8h*)p;
  v8h hi = *(const v8h*)(p + 8);
  v16h r;
#pragma unroll
  for (int i = 0; i < 8; ++i) { r[i] = lo[i]; r[i + 8] = hi[i]; }
  return r;
}

// B-operand tile (32 K x 16 cols) where column n of B is row n of the
// row-major source (B = source^T). Lane l: col = l&15; K = k0+(l>>4)*16+0..15.
__device__ __forceinline__ v16h load_b_tile(const _Float16* __restrict__ base,
                                            int stride, int k0) {
  const int lane = threadIdx.x & 31;
  const _Float16* p = base + (size_t)(lane & 15) * stride + k0 + ((lane >> 4) << 4);
  v8h lo = *(const v8h*)p;
  v8h hi = *(const v8h*)(p + 8);
  v16h r;
#pragma unroll
  for (int i = 0; i < 8; ++i) { r[i] = lo[i]; r[i + 8] = hi[i]; }
  return r;
}

// Exchange value with lane l^16 (ds_swizzle SWAPX16: xor=0x10, and=0x1f).
__device__ __forceinline__ float swap16f(float v) {
  return __int_as_float(__builtin_amdgcn_ds_swizzle(__float_as_int(v), 0x401F));
}
__device__ __forceinline__ int swap16i(int v) {
  return __builtin_amdgcn_ds_swizzle(v, 0x401F);
}

// 16B async copy global -> LDS (ASYNCcnt-tracked); sync fallback otherwise.
__device__ __forceinline__ void copy_b128_to_lds(const _Float16* g, _Float16* l) {
#if HAVE_ASYNC_LDS
  __builtin_amdgcn_global_load_async_to_lds_b128(
      (AS_GLOBAL v4i*)(void*)g, (AS_LOCAL v4i*)(void*)l, 0, 0);
#else
  v8h t = *(const v8h*)g;
  *(v8h*)l = t;
#endif
}

__device__ __forceinline__ void stage_pair(const _Float16* gk, const _Float16* gv,
                                           _Float16* lk, _Float16* lv) {
  copy_b128_to_lds(gk, lk);
  copy_b128_to_lds(gv, lv);
}

__device__ __forceinline__ void wait_async() {
#if HAVE_ASYNC_LDS
#if __has_builtin(__builtin_amdgcn_s_wait_asynccnt)
  __builtin_amdgcn_s_wait_asynccnt(0);
#else
  asm volatile("s_wait_asynccnt 0x0" ::: "memory");
#endif
#endif
}

__global__ void cvt_f32_f16(const float* __restrict__ s, _Float16* __restrict__ d,
                            int n) {
  int i = blockIdx.x * blockDim.x + threadIdx.x;
  if (i < n) d[i] = (_Float16)s[i];
}

// y = A(MxK f16) @ W^T(NxK f16 row-major) + bias, 32x64 strip per wave,
// register double-buffered A operands.
// mode 0: f16 -> Qh[h][t][d] scaled by 0.125*log2(e);  mode 1: f16 -> Kh[h][t][d]
// mode 2: f16 -> Vt[h][d][t] (transposed);             mode 3: f32 -> outF[t][n]
__global__ void __launch_bounds__(256)
gemm32_wmma(const _Float16* __restrict__ A, const _Float16* __restrict__ W,
            const float* __restrict__ bias, _Float16* __restrict__ outH,
            float* __restrict__ outF, int M, int N, int K, int mode) {
  const int wid = (int)((blockIdx.x * blockDim.x + threadIdx.x) >> 5);
  const int nstrips = N / (16 * NSUB);
  const int total = (M / (16 * MSUB)) * nstrips;
  if (wid >= total) return;
  const int mt = wid / nstrips;
  const int ns = wid % nstrips;
  const _Float16* A0 = A + (size_t)(mt * 32) * K;
  const _Float16* A1 = A0 + (size_t)16 * K;

  v8f acc0[NSUB] = {};
  v8f acc1[NSUB] = {};
  v16h a0 = load_a_tile(A0, K, 0);
  v16h a1 = load_a_tile(A1, K, 0);

  for (int k0 = 0; k0 < K; k0 += 32) {
    v16h b[NSUB];
#pragma unroll
    for (int j = 0; j < NSUB; ++j) {
      const _Float16* Wrow = W + (size_t)((ns * NSUB + j) << 4) * K;
      b[j] = load_b_tile(Wrow, K, k0);
    }
    v16h a0n = a0, a1n = a1;
    if (k0 + 32 < K) {  // preload next A tiles under the WMMAs
      a0n = load_a_tile(A0, K, k0 + 32);
      a1n = load_a_tile(A1, K, k0 + 32);
    }
#pragma unroll
    for (int j = 0; j < NSUB; ++j) acc0[j] = wmma_f16(a0, b[j], acc0[j]);
#pragma unroll
    for (int j = 0; j < NSUB; ++j) acc1[j] = wmma_f16(a1, b[j], acc1[j]);
    a0 = a0n; a1 = a1n;
  }

  const int lane = threadIdx.x & 31;
  const int rbase = (lane >> 4) << 3;  // C/D layout: M = r + 8*(lane>=16)
#pragma unroll
  for (int sub = 0; sub < MSUB; ++sub) {
#pragma unroll
    for (int j = 0; j < NSUB; ++j) {
      const int n = ((ns * NSUB + j) << 4) + (lane & 15);
      const float bv = bias[n];
#pragma unroll
      for (int r = 0; r < 8; ++r) {
        const int mrow = mt * 32 + sub * 16 + r + rbase;
        float val = (sub ? acc1[j][r] : acc0[j][r]) + bv;
        if (mode == 0) {
          val *= 0.125f * 1.44269504088896340736f;  // 1/sqrt(64) * log2(e)
          outH[(size_t)(n >> 6) * T_SEQ * D_HEAD + (size_t)mrow * D_HEAD + (n & 63)] =
              (_Float16)val;
        } else if (mode == 1) {
          outH[(size_t)(n >> 6) * T_SEQ * D_HEAD + (size_t)mrow * D_HEAD + (n & 63)] =
              (_Float16)val;
        } else if (mode == 2) {
          outH[(size_t)(n >> 6) * D_HEAD * T_SEQ + (size_t)(n & 63) * T_SEQ + mrow] =
              (_Float16)val;
        } else {
          outF[(size_t)mrow * N + n] = val;
        }
      }
    }
  }
}

// Cooperative flash attention: 8 waves/WG share K and V^T tiles staged in LDS
// (double-buffered, async-to-LDS when available). Each wave owns one 16-query
// block; computes S^T = K*Q^T (softmax reductions in-lane + one SWAPX16),
// then O^T += V^T * P^T.
__global__ void __launch_bounds__(256)
attn_fa_wmma(const _Float16* __restrict__ Qh, const _Float16* __restrict__ Kh,
             const _Float16* __restrict__ Vt, _Float16* __restrict__ ctx) {
  __shared__ __align__(16) _Float16 skK[2][32 * 64];  // 32 keys x 64 d
  __shared__ __align__(16) _Float16 skV[2][64 * 32];  // 64 d x 32 keys (V^T)

  const int wgid = blockIdx.x;                 // 384 WGs = 12 heads x 32
  const int h = wgid >> 5;
  const int qb = ((wgid & 31) << 3) + (threadIdx.x >> 5);  // q-block 0..255
  const int lane = threadIdx.x & 31;
  const int tid = threadIdx.x;
  const bool hihalf = (lane & 16) != 0;

  const _Float16* Qbase = Qh + (size_t)h * T_SEQ * D_HEAD + (size_t)(qb << 4) * D_HEAD;
  const _Float16* Kbase = Kh + (size_t)h * T_SEQ * D_HEAD;
  const _Float16* Vbase = Vt + (size_t)h * D_HEAD * T_SEQ;

  // Per-thread staging addresses: 16B each of K block (linear) and V^T block.
  const _Float16* gK0 = Kbase + (size_t)tid * 8;              // + kb*64
  const int vd = tid >> 2, vp = tid & 3;
  const _Float16* gV0 = Vbase + (size_t)vd * T_SEQ + vp * 8;  // + kb
  _Float16* dK[2] = {&skK[0][tid * 8], &skK[1][tid * 8]};
  _Float16* dV[2] = {&skV[0][vd * 32 + vp * 8], &skV[1][vd * 32 + vp * 8]};

  // Q^T B-operands, resident for the whole key loop (d = 0..31, 32..63)
  v16h qlo = load_b_tile(Qbase, D_HEAD, 0);
  v16h qhi = load_b_tile(Qbase, D_HEAD, 32);

  float mrun = -3.0e38f;
  float lrun = 0.0f;
  v8f acc[4] = {};  // O^T: 4 d-tiles x (16d x 16q)

  stage_pair(gK0, gV0, dK[0], dV[0]);  // prologue: stage kb=0 into buffer 0

  for (int kb = 0; kb < T_SEQ; kb += 32) {
    const int b = (kb >> 5) & 1;
    wait_async();
    __syncthreads();  // buffer b complete & visible; buffer b^1 free
    if (kb + 32 < T_SEQ) {
      stage_pair(gK0 + (size_t)(kb + 32) * 64, gV0 + (kb + 32),
                 dK[b ^ 1], dV[b ^ 1]);  // overlaps with consume below
    }

    // S^T tiles from LDS K block: rows = keys, cols = queries, K-dim = d
    v8f s0 = {}, s1 = {};
    s0 = wmma_f16(load_a_tile(&skK[b][0], D_HEAD, 0),  qlo, s0);
    s0 = wmma_f16(load_a_tile(&skK[b][0], D_HEAD, 32), qhi, s0);
    s1 = wmma_f16(load_a_tile(&skK[b][16 * 64], D_HEAD, 0),  qlo, s1);
    s1 = wmma_f16(load_a_tile(&skK[b][16 * 64], D_HEAD, 32), qhi, s1);

    // Online softmax (scores pre-scaled by 0.125*log2e -> exp2)
    float pm = s0[0];
#pragma unroll
    for (int i = 1; i < 8; ++i) pm = fmaxf(pm, s0[i]);
#pragma unroll
    for (int i = 0; i < 8; ++i) pm = fmaxf(pm, s1[i]);
    pm = fmaxf(pm, swap16f(pm));
    const float mnew = fmaxf(mrun, pm);
    const float alpha = exp2f(mrun - mnew);

    float p0[8], p1[8];
    float ps = 0.0f;
#pragma unroll
    for (int i = 0; i < 8; ++i) { p0[i] = exp2f(s0[i] - mnew); ps += p0[i]; }
#pragma unroll
    for (int i = 0; i < 8; ++i) { p1[i] = exp2f(s1[i] - mnew); ps += p1[i]; }
    ps += swap16f(ps);
    lrun = lrun * alpha + ps;
    mrun = mnew;
#pragma unroll
    for (int j = 0; j < 4; ++j) {
#pragma unroll
      for (int i = 0; i < 8; ++i) acc[j][i] *= alpha;
    }

    // Assemble P^T B-operand (32 keys x 16 queries):
    // lane<16 -> tile0 keys 0..15 (own p0 | partner p0),
    // lane>=16 -> tile1 keys 16..31 (partner p1 | own p1).
    union { v16h v; int u[8]; } pb;
#pragma unroll
    for (int i = 0; i < 4; ++i) {
      union { _Float16 h2[2]; int i32; } x, y;
      x.h2[0] = (_Float16)p0[2 * i]; x.h2[1] = (_Float16)p0[2 * i + 1];
      y.h2[0] = (_Float16)p1[2 * i]; y.h2[1] = (_Float16)p1[2 * i + 1];
      const int xs = swap16i(x.i32);
      const int ys = swap16i(y.i32);
      pb.u[i]     = hihalf ? ys : x.i32;
      pb.u[i + 4] = hihalf ? y.i32 : xs;
    }

    // O^T += V^T * P^T  (A-operand = V^T tiles from LDS, 16d x 32keys)
#pragma unroll
    for (int dt = 0; dt < 4; ++dt) {
      acc[dt] = wmma_f16(load_a_tile(&skV[b][dt * 16 * 32], 32, 0), pb.v, acc[dt]);
    }
  }

  // Epilogue: O^T lane layout -> ctx[t][h*64+d] (f16 feed for the Wo GEMM)
  const float inv = 1.0f / lrun;
  const int q = (qb << 4) + (lane & 15);
  const int rb = (lane >> 4) << 3;
#pragma unroll
  for (int dt = 0; dt < 4; ++dt) {
#pragma unroll
    for (int r = 0; r < 8; ++r) {
      const int d = (dt << 4) + r + rb;
      ctx[(size_t)q * C_DIM + h * D_HEAD + d] = (_Float16)(acc[dt][r] * inv);
    }
  }
}

extern "C" void kernel_launch(void* const* d_in, const int* in_sizes, int n_in,
                              void* d_out, int out_size, void* d_ws, size_t ws_size,
                              hipStream_t stream) {
  (void)in_sizes; (void)n_in; (void)out_size; (void)ws_size;
  const float* x  = (const float*)d_in[0];
  const float* Wq = (const float*)d_in[1];
  const float* bq = (const float*)d_in[2];
  const float* Wk = (const float*)d_in[3];
  const float* bk = (const float*)d_in[4];
  const float* Wv = (const float*)d_in[5];
  const float* bv = (const float*)d_in[6];
  const float* Wo = (const float*)d_in[7];
  const float* bo = (const float*)d_in[8];

  const size_t TC = (size_t)T_SEQ * C_DIM;   // 3,145,728
  const size_t CC = (size_t)C_DIM * C_DIM;   // 589,824
  _Float16* p = (_Float16*)d_ws;
  _Float16* x_h  = p; p += TC;
  _Float16* wq_h = p; p += CC;
  _Float16* wk_h = p; p += CC;
  _Float16* wv_h = p; p += CC;
  _Float16* wo_h = p; p += CC;
  _Float16* Qh   = p; p += TC;  // [H,T,D], pre-scaled by 0.125*log2(e)
  _Float16* Kh   = p; p += TC;  // [H,T,D]
  _Float16* Vt   = p; p += TC;  // [H,D,T]
  _Float16* ctx  = p; p += TC;  // [T,C]

  cvt_f32_f16<<<dim3((unsigned)((TC + 255) / 256)), 256, 0, stream>>>(x, x_h, (int)TC);
  cvt_f32_f16<<<dim3((unsigned)((CC + 255) / 256)), 256, 0, stream>>>(Wq, wq_h, (int)CC);
  cvt_f32_f16<<<dim3((unsigned)((CC + 255) / 256)), 256, 0, stream>>>(Wk, wk_h, (int)CC);
  cvt_f32_f16<<<dim3((unsigned)((CC + 255) / 256)), 256, 0, stream>>>(Wv, wv_h, (int)CC);
  cvt_f32_f16<<<dim3((unsigned)((CC + 255) / 256)), 256, 0, stream>>>(Wo, wo_h, (int)CC);

  // Projections: (T/32)*(C/64) = 1536 waves, 8 waves (256 thr) per block
  const int gwaves = (T_SEQ / (16 * MSUB)) * (C_DIM / (16 * NSUB));
  dim3 gblocks((unsigned)(gwaves / 8));
  gemm32_wmma<<<gblocks, 256, 0, stream>>>(x_h, wq_h, bq, Qh, nullptr,
                                           T_SEQ, C_DIM, C_DIM, 0);
  gemm32_wmma<<<gblocks, 256, 0, stream>>>(x_h, wk_h, bk, Kh, nullptr,
                                           T_SEQ, C_DIM, C_DIM, 1);
  gemm32_wmma<<<gblocks, 256, 0, stream>>>(x_h, wv_h, bv, Vt, nullptr,
                                           T_SEQ, C_DIM, C_DIM, 2);

  // Attention: 12 heads x 32 WGs, 8 waves per WG (128 queries/WG)
  attn_fa_wmma<<<dim3(H_HEADS * 32), 256, 0, stream>>>(Qh, Kh, Vt, ctx);

  // Output projection -> f32 d_out
  gemm32_wmma<<<gblocks, 256, 0, stream>>>(ctx, wo_h, bo, nullptr, (float*)d_out,
                                           T_SEQ, C_DIM, C_DIM, 3);
}
